// Recurrence_59493886984360
// MI455X (gfx1250) — matrix operand order repaired and checked by reading, not compiled, not run on hardware
//
#include <hip/hip_runtime.h>

// ---------------------------------------------------------------------------
// Types for CDNA5 WMMA (gfx1250, wave32): v_wmma_f32_16x16x32_bf16
// ---------------------------------------------------------------------------
typedef __attribute__((ext_vector_type(16))) __bf16 v16bf;
typedef __attribute__((ext_vector_type(8)))  float  v8f;

union BfPack { uint4 u[2]; v16bf v; };

static constexpr int H     = 256;
static constexpr int L     = 64;
static constexpr int T     = 64;
static constexpr int NB    = 64;                       // batch
static constexpr int STATE = 1 + L + 1 + H + L * H;    // 16706
static constexpr int NL    = NB * L;                   // 4096

static __device__ __forceinline__ float sigm(float x) {
  return 1.0f / (1.0f + expf(-x));
}

// ---------------------------------------------------------------------------
// Register-blocked WMMA GEMM:  C[M x N] = act( A[M x K] * W[N x K]^T + bias )
//   A: f32 (converted in-register to bf16) or bf16, row-major, K % 32 == 0,
//      M % 16 == 0.  W: bf16 row-major (out,in) = the reference weight layout
//      (x @ W.T).  C: f32 row-major, ld = N, N edges guarded per 16-col tile.
// Each wave owns a 16x64 output strip (4 accumulators): the A fragment is
// loaded once per K-chunk and reused by 4 back-to-back v_wmma ops against 4
// streamed B fragments -> 4x fewer A loads, 4 independent WMMAs to pipeline.
// Fragment packing per CDNA5 ISA 7.12.2:
//   A lane m=lane&15; e=0..7 -> k=kbase+e (kbase=kc+8*hi), e=8..15 -> k=kbase+16+(e-8)
//   B lane n=lane&15; same k mapping as A (kbase=kc+8*hi)
//   C/D lane: m = 16*tM + v + 8*hi, n = ncol_tile + (lane&15)
// ---------------------------------------------------------------------------
template <bool AF32>
__global__ void __launch_bounds__(256)
k_gemm(const void* __restrict__ Ap, const __bf16* __restrict__ W,
       const float* __restrict__ bias, float* __restrict__ C,
       int M, int N, int K, int act)
{
  const int lane    = threadIdx.x & 31;
  const int wav     = threadIdx.x >> 5;
  const int tilesNg = (N + 63) >> 6;               // 64-wide column groups
  const int tile    = blockIdx.x * 8 + wav;
  const int tM      = tile / tilesNg;
  const int tG      = tile - tM * tilesNg;
  if (tM * 16 >= M) return;

  const int hi   = lane >> 4;     // 0 or 1
  const int l16  = lane & 15;
  const int mrow = tM * 16 + l16;
  const int ncg  = tG * 64 + l16; // this lane's column in group tile 0

  v8f acc[4];
#pragma unroll
  for (int j = 0; j < 4; ++j) acc[j] = (v8f){0.f,0.f,0.f,0.f,0.f,0.f,0.f,0.f};

  const float*  Af = (const float*)Ap;
  const __bf16* Ab = (const __bf16*)Ap;

  for (int kc = 0; kc < K; kc += 32) {
    const int ka = kc + hi * 8;
    v16bf a;
    if (AF32) {
      const float* ap = Af + (size_t)mrow * K + ka;
      float4 f0 = *(const float4*)(ap);
      float4 f1 = *(const float4*)(ap + 4);
      float4 f2 = *(const float4*)(ap + 16);
      float4 f3 = *(const float4*)(ap + 20);
      a[0]  = (__bf16)f0.x; a[1]  = (__bf16)f0.y; a[2]  = (__bf16)f0.z; a[3]  = (__bf16)f0.w;
      a[4]  = (__bf16)f1.x; a[5]  = (__bf16)f1.y; a[6]  = (__bf16)f1.z; a[7]  = (__bf16)f1.w;
      a[8]  = (__bf16)f2.x; a[9]  = (__bf16)f2.y; a[10] = (__bf16)f2.z; a[11] = (__bf16)f2.w;
      a[12] = (__bf16)f3.x; a[13] = (__bf16)f3.y; a[14] = (__bf16)f3.z; a[15] = (__bf16)f3.w;
      if (kc + 32 < K) __builtin_prefetch((const void*)(ap + 32), 0, 0);
    } else {
      const __bf16* ap = Ab + (size_t)mrow * K + ka;
      BfPack pa;
      pa.u[0] = *(const uint4*)(ap);        // k = ka .. ka+7
      pa.u[1] = *(const uint4*)(ap + 16);   // k = ka+16 .. ka+23
      a = pa.v;
      if (kc + 32 < K) __builtin_prefetch((const void*)(ap + 32), 0, 0);
    }

    const int kb = kc + hi * 8;
#pragma unroll
    for (int j = 0; j < 4; ++j) {
      const int nc = ncg + j * 16;
      v16bf b;
      if (nc < N) {
        const __bf16* bp = W + (size_t)nc * K + kb;
        BfPack pb;
        pb.u[0] = *(const uint4*)(bp);      // k = kb .. kb+7
        pb.u[1] = *(const uint4*)(bp + 16); // k = kb+16 .. kb+23
        b = pb.v;
      } else {
        BfPack pb;
        pb.u[0] = make_uint4(0, 0, 0, 0);
        pb.u[1] = make_uint4(0, 0, 0, 0);
        b = pb.v;
      }
      acc[j] = __builtin_amdgcn_wmma_f32_16x16x32_bf16(
          /*neg_a=*/false, a, /*neg_b=*/false, b,
          /*c_mod=*/(short)0, acc[j], /*reuse_a=*/false, /*reuse_b=*/false);
    }
  }

#pragma unroll
  for (int j = 0; j < 4; ++j) {
    const int nc = ncg + j * 16;
    if (nc < N) {
      const float bb = bias ? bias[nc] : 0.0f;
#pragma unroll
      for (int v = 0; v < 8; ++v) {
        float val = acc[j][v] + bb;
        if (act == 1) val = fmaxf(val, 0.0f);
        C[(size_t)(tM * 16 + v + hi * 8) * N + nc] = val;
      }
    }
  }
}

// ---------------------------------------------------------------------------
// Small helper kernels
// ---------------------------------------------------------------------------
__global__ void k_cvt(const float* __restrict__ s, __bf16* __restrict__ d, int n) {
  int i = blockIdx.x * blockDim.x + threadIdx.x;
  if (i < n) d[i] = (__bf16)s[i];
}

__global__ void k_cvt_pad(const float* __restrict__ s, __bf16* __restrict__ d,
                          int rows, int incols, int outcols) {
  int i = blockIdx.x * blockDim.x + threadIdx.x;
  if (i >= rows * outcols) return;
  int r = i / outcols, c = i - r * outcols;
  d[i] = (c < incols) ? (__bf16)s[(size_t)r * incols + c] : (__bf16)0.0f;
}

__global__ void k_zerof(float* __restrict__ p, int n) {
  int i = blockIdx.x * blockDim.x + threadIdx.x;
  if (i < n) p[i] = 0.0f;
}

__global__ void k_zeroint(int* p) { if (threadIdx.x == 0) *p = 0; }

__global__ void k_anynz(const float* __restrict__ x, int n, int* __restrict__ f) {
  int i = blockIdx.x * blockDim.x + threadIdx.x;
  if (i < n && x[i] != 0.0f) atomicOr(f, 1);
}

// Conv layer 1 (cin = 1) on rewards[0]; output channel-last (n*64+p, co)
__global__ void __launch_bounds__(256)
k_conv1(const float* __restrict__ in, const float* __restrict__ Wc,
        const float* __restrict__ bc, float* __restrict__ out)
{
  int n = blockIdx.x, co = threadIdx.x;
  __shared__ float img[64];
  if (co < 64) img[co] = in[n * 66 + co];
  __syncthreads();
  float w[9];
#pragma unroll
  for (int j = 0; j < 9; ++j) w[j] = Wc[co * 9 + j];
  const float bb = bc[co];
  for (int p = 0; p < 64; ++p) {
    int py = p >> 3, px = p & 7;
    float acc = bb;
#pragma unroll
    for (int ky = 0; ky < 3; ++ky)
#pragma unroll
      for (int kx = 0; kx < 3; ++kx) {
        int iy = py + ky - 1, ix = px + kx - 1;
        if (iy >= 0 && iy < 8 && ix >= 0 && ix < 8)
          acc += img[iy * 8 + ix] * w[ky * 3 + kx];
      }
    out[(size_t)(n * 64 + p) * 256 + co] = fmaxf(acc, 0.0f);
  }
}

// im2col for channel-last (n*64+p, ci) 8x8 SAME 3x3 -> bf16 col (row, ci*9+kid)
__global__ void __launch_bounds__(256)
k_im2col(const float* __restrict__ x, __bf16* __restrict__ col)
{
  int row = blockIdx.x;          // n*64 + p
  int ci  = threadIdx.x;         // 0..255
  int p = row & 63, py = p >> 3, px = p & 7, nb = row & ~63;
#pragma unroll
  for (int kid = 0; kid < 9; ++kid) {
    int ky = kid / 3, kx = kid - ky * 3;
    int iy = py + ky - 1, ix = px + kx - 1;
    float v = 0.0f;
    if (iy >= 0 && iy < 8 && ix >= 0 && ix < 8)
      v = x[(size_t)(nb + iy * 8 + ix) * 256 + ci];
    col[(size_t)row * 2304 + ci * 9 + kid] = (__bf16)v;
  }
}

// M_init = new_ep ? M0 : rnn_hxs memory slice   (layout (n,l,h))
__global__ void k_selectM(const float* __restrict__ rnn, const float* __restrict__ M0,
                          const int* __restrict__ flag, float* __restrict__ Mb)
{
  int i = blockIdx.x * blockDim.x + threadIdx.x;
  if (i >= NB * L * H) return;
  int n = i >> 14, r = i & 16383;
  Mb[i] = (*flag) ? rnn[(size_t)n * STATE + 322 + r] : M0[i];
}

__global__ void k_copyh(const float* __restrict__ rnn, float* __restrict__ h) {
  int i = blockIdx.x * blockDim.x + threadIdx.x;   // 64*256
  h[i] = rnn[(size_t)(i >> 8) * STATE + 66 + (i & 255)];
}

// Encoder GRU combine: h' = (1-z)*tanh(inn + r*hn) + z*h ; write K slot
__global__ void __launch_bounds__(256)
k_enc_combine(const float* __restrict__ gi, const float* __restrict__ gh,
              float* __restrict__ h, float* __restrict__ Kb, int l, int dir)
{
  int idx = blockIdx.x * blockDim.x + threadIdx.x;   // 64*256
  int n = idx >> 8, d = idx & 255;
  const float* gir = gi + (size_t)(n * 64 + l) * 768;
  const float* ghr = gh + (size_t)n * 768;
  float r  = sigm(gir[d] + ghr[d]);
  float z  = sigm(gir[256 + d] + ghr[256 + d]);
  float nn = tanhf(gir[512 + d] + r * ghr[512 + d]);
  float hn = (1.0f - z) * nn + z * h[idx];
  h[idx] = hn;
  Kb[(size_t)(n * 64 + l) * 512 + dir * 256 + d] = hn;
}

// Attention: wk = softmax(K.qk), wm = softmax(M.qm), probs -> out[1..64], a_t -> out[0]
__global__ void __launch_bounds__(256)
k_attn(const float* __restrict__ g, const float* __restrict__ Kb,
       const float* __restrict__ Mb, const float* __restrict__ in_t,
       float* __restrict__ out_t)
{
  int n = blockIdx.x, tid = threadIdx.x;
  __shared__ float sq[768];
  __shared__ float sred[256];
  __shared__ float sWk[64], sWm[64];
  __shared__ float sstat[4];
  for (int i = tid; i < 768; i += 256) sq[i] = g[(size_t)n * 769 + i];
  __syncthreads();
  int l = tid >> 2, q = tid & 3;
  const float* Krow = Kb + (size_t)(n * 64 + l) * 512 + q * 128;
  float acc = 0.0f;
  for (int i = 0; i < 128; ++i) acc += Krow[i] * sq[q * 128 + i];
  sred[tid] = acc;
  __syncthreads();
  if (q == 0) sWk[l] = sred[tid] + sred[tid + 1] + sred[tid + 2] + sred[tid + 3];
  __syncthreads();
  const float* Mrow = Mb + (size_t)(n * 64 + l) * 256 + q * 64;
  acc = 0.0f;
  for (int i = 0; i < 64; ++i) acc += Mrow[i] * sq[512 + q * 64 + i];
  sred[tid] = acc;
  __syncthreads();
  if (q == 0) sWm[l] = sred[tid] + sred[tid + 1] + sred[tid + 2] + sred[tid + 3];
  __syncthreads();
  if (tid == 0) {
    float mk = -1e30f, mm = -1e30f;
    for (int i = 0; i < 64; ++i) { mk = fmaxf(mk, sWk[i]); mm = fmaxf(mm, sWm[i]); }
    sstat[0] = mk; sstat[2] = mm;
  }
  __syncthreads();
  if (tid < 64) { sWk[tid] = expf(sWk[tid] - sstat[0]); sWm[tid] = expf(sWm[tid] - sstat[2]); }
  __syncthreads();
  if (tid == 0) {
    float sk = 0.0f, sm = 0.0f;
    for (int i = 0; i < 64; ++i) { sk += sWk[i]; sm += sWm[i]; }
    sstat[1] = sk; sstat[3] = sm;
  }
  __syncthreads();
  float bs = sigm(g[(size_t)n * 769 + 768]);
  if (tid < 64)
    out_t[(size_t)n * STATE + 1 + tid] =
        bs * sWk[tid] / sstat[1] + (1.0f - bs) * sWm[tid] / sstat[3];
  if (tid == 0) out_t[(size_t)n * STATE] = in_t[n * 66 + 65];
}

// y = [M[n, a_t], go_t, 0-pad to 288]
__global__ void k_gather(const float* __restrict__ Mb, const float* __restrict__ in_t,
                         float* __restrict__ y)
{
  int n = blockIdx.x, j = threadIdx.x;   // 288 threads
  int a = (int)in_t[n * 66 + 65];
  float v = 0.0f;
  if (j < 256)       v = Mb[(size_t)(n * 64 + a) * 256 + j];
  else if (j == 256) v = in_t[n * 66 + 64];
  y[(size_t)n * 288 + j] = v;
}

// Main GRU combine; writes new h and out[66..321]
__global__ void __launch_bounds__(256)
k_gru_main(const float* __restrict__ gi, const float* __restrict__ gh,
           float* __restrict__ h, float* __restrict__ out_t)
{
  int idx = blockIdx.x * blockDim.x + threadIdx.x;   // 64*256
  int n = idx >> 8, d = idx & 255;
  const float* gir = gi + (size_t)n * 768;
  const float* ghr = gh + (size_t)n * 768;
  float r  = sigm(gir[d] + ghr[d]);
  float z  = sigm(gir[256 + d] + ghr[256 + d]);
  float nn = tanhf(gir[512 + d] + r * ghr[512 + d]);
  float hn = (1.0f - z) * nn + z * h[idx];
  h[idx] = hn;
  out_t[(size_t)n * STATE + 66 + d] = hn;
}

__global__ void k_vout(const float* __restrict__ v, float* __restrict__ out_t) {
  int n = threadIdx.x;
  if (n < 64) out_t[(size_t)n * STATE + 65] = v[n];
}

// Memory write (one-hot row) + stream full M into out[322..]
__global__ void __launch_bounds__(256)
k_emitM(float* __restrict__ Mb, const float* __restrict__ wr,
        const float* __restrict__ in_t, float* __restrict__ out_t)
{
  int idx = blockIdx.x * blockDim.x + threadIdx.x;   // 64*64*64 float4 units
  int n = idx >> 12, rem = idx & 4095, l = rem >> 6, d4 = (rem & 63) << 2;
  int a = (int)in_t[n * 66 + 65];
  float4 mv = *(const float4*)(Mb + ((size_t)(n * 64 + l) << 8) + d4);
  if (l == a) {
    const float* wn = wr + (size_t)n * 257;
    float s = sigm(wn[256]);
    mv.x = s * wn[d4 + 0] + (1.0f - s) * mv.x;
    mv.y = s * wn[d4 + 1] + (1.0f - s) * mv.y;
    mv.z = s * wn[d4 + 2] + (1.0f - s) * mv.z;
    mv.w = s * wn[d4 + 3] + (1.0f - s) * mv.w;
    *(float4*)(Mb + ((size_t)(n * 64 + l) << 8) + d4) = mv;
  }
  float* o = out_t + (size_t)n * STATE + 322 + (l << 8) + d4;  // STATE even, not /4
  o[0] = mv.x; o[1] = mv.y; o[2] = mv.z; o[3] = mv.w;
}

__global__ void k_lastcopy(float* __restrict__ out) {
  int i = blockIdx.x * blockDim.x + threadIdx.x;
  if (i < NB * STATE)
    out[(size_t)T * NB * STATE + i] = out[(size_t)(T - 1) * NB * STATE + i];
}

// ---------------------------------------------------------------------------
// Host orchestration
// ---------------------------------------------------------------------------
extern "C" void kernel_launch(void* const* d_in, const int* in_sizes, int n_in,
                              void* d_out, int out_size, void* d_ws, size_t ws_size,
                              hipStream_t stream)
{
  (void)in_sizes; (void)n_in; (void)out_size; (void)ws_size;

  // Input order: top-level dict insertion order; nested dicts flattened in
  // sorted-key order (Whh, Wih, bhh, bih), tuples/lists in order.
  const float* inp   = (const float*)d_in[0];
  const float* hxs   = (const float*)d_in[1];
  const float* cW1 = (const float*)d_in[2];  const float* cb1 = (const float*)d_in[3];
  const float* cW2 = (const float*)d_in[4];  const float* cb2 = (const float*)d_in[5];
  const float* cW3 = (const float*)d_in[6];  const float* cb3 = (const float*)d_in[7];
  const float* f0W = (const float*)d_in[8];  const float* f0b = (const float*)d_in[9];
  const float* f1W = (const float*)d_in[10]; const float* f1b = (const float*)d_in[11];
  const float* f2W = (const float*)d_in[12]; const float* f2b = (const float*)d_in[13];
  const float* e0Whh = (const float*)d_in[14]; const float* e0Wih = (const float*)d_in[15];
  const float* e0bhh = (const float*)d_in[16]; const float* e0bih = (const float*)d_in[17];
  const float* e1Whh = (const float*)d_in[18]; const float* e1Wih = (const float*)d_in[19];
  const float* e1bhh = (const float*)d_in[20]; const float* e1bih = (const float*)d_in[21];
  const float* gWhh = (const float*)d_in[22]; const float* gWih = (const float*)d_in[23];
  const float* gbhh = (const float*)d_in[24]; const float* gbih = (const float*)d_in[25];
  const float* goW  = (const float*)d_in[26]; const float* gob  = (const float*)d_in[27];
  const float* wrW  = (const float*)d_in[28]; const float* wrbias = (const float*)d_in[29];
  const float* crW  = (const float*)d_in[30]; const float* crb  = (const float*)d_in[31];
  float* out = (float*)d_out;

  char* wsp = (char*)d_ws;
  size_t off = 0;
  auto alloc = [&](size_t bytes) -> void* {
    void* p = wsp + off;
    off = (off + bytes + 255) & ~(size_t)255;
    return p;
  };

  // bf16 weights (converted once per call)
  __bf16* wc2b   = (__bf16*)alloc((size_t)256 * 2304 * 2);
  __bf16* wc3b   = (__bf16*)alloc((size_t)256 * 2304 * 2);
  __bf16* eWih0b = (__bf16*)alloc((size_t)768 * 256 * 2);
  __bf16* eWhh0b = (__bf16*)alloc((size_t)768 * 256 * 2);
  __bf16* eWih1b = (__bf16*)alloc((size_t)768 * 256 * 2);
  __bf16* eWhh1b = (__bf16*)alloc((size_t)768 * 256 * 2);
  __bf16* gWihb  = (__bf16*)alloc((size_t)768 * 256 * 2);
  __bf16* gWhhb  = (__bf16*)alloc((size_t)768 * 256 * 2);
  __bf16* goWb   = (__bf16*)alloc((size_t)769 * 256 * 2);
  __bf16* wrWb   = (__bf16*)alloc((size_t)257 * 256 * 2);
  __bf16* crWb   = (__bf16*)alloc((size_t)256 * 2);
  __bf16* f0Wb   = (__bf16*)alloc((size_t)256 * 288 * 2);   // padded 257->288
  __bf16* f1Wb   = (__bf16*)alloc((size_t)256 * 256 * 2);
  __bf16* f2Wb   = (__bf16*)alloc((size_t)256 * 256 * 2);
  // activations / state
  float*  x1   = (float*)alloc((size_t)NL * 256 * 4);
  __bf16* colb = (__bf16*)alloc((size_t)NL * 2304 * 2);
  float*  x2   = (float*)alloc((size_t)NL * 256 * 4);
  float*  M0f  = (float*)alloc((size_t)NL * 256 * 4);       // (n,l,h)
  float*  giF  = (float*)alloc((size_t)NL * 768 * 4);
  float*  giB  = (float*)alloc((size_t)NL * 768 * 4);
  float*  henc0 = (float*)alloc((size_t)64 * 256 * 4);
  float*  henc1 = (float*)alloc((size_t)64 * 256 * 4);
  float*  ghE0  = (float*)alloc((size_t)64 * 768 * 4);
  float*  ghE1  = (float*)alloc((size_t)64 * 768 * 4);
  float*  Kbuf  = (float*)alloc((size_t)NL * 512 * 4);      // (n,l,2H)
  float*  h     = (float*)alloc((size_t)64 * 256 * 4);
  float*  Mb    = (float*)alloc((size_t)NL * 256 * 4);
  float*  gbuf  = (float*)alloc((size_t)64 * 769 * 4);
  float*  ybuf  = (float*)alloc((size_t)64 * 288 * 4);
  float*  y1    = (float*)alloc((size_t)64 * 256 * 4);
  float*  y2    = (float*)alloc((size_t)64 * 256 * 4);
  float*  y3    = (float*)alloc((size_t)64 * 256 * 4);
  float*  gib   = (float*)alloc((size_t)64 * 768 * 4);
  float*  ghb   = (float*)alloc((size_t)64 * 768 * 4);
  float*  wrD   = (float*)alloc((size_t)64 * 257 * 4);
  float*  vb    = (float*)alloc((size_t)64 * 4);
  int*    flag  = (int*)alloc(256);

  auto cvt = [&](const float* s, __bf16* d, int n) {
    k_cvt<<<(n + 255) / 256, 256, 0, stream>>>(s, d, n);
  };
  auto gemmF = [&](const float* A, const __bf16* W, const float* bias, float* C,
                   int M, int N, int K, int act) {
    int tiles = ((M + 15) / 16) * ((N + 63) / 64);
    k_gemm<true><<<(tiles + 7) / 8, 256, 0, stream>>>(A, W, bias, C, M, N, K, act);
  };
  auto gemmB = [&](const __bf16* A, const __bf16* W, const float* bias, float* C,
                   int M, int N, int K, int act) {
    int tiles = ((M + 15) / 16) * ((N + 63) / 64);
    k_gemm<false><<<(tiles + 7) / 8, 256, 0, stream>>>(A, W, bias, C, M, N, K, act);
  };

  // --- weight conversion (f32 -> bf16) ---
  cvt(cW2, wc2b, 256 * 2304);
  cvt(cW3, wc3b, 256 * 2304);
  cvt(e0Wih, eWih0b, 768 * 256);  cvt(e0Whh, eWhh0b, 768 * 256);
  cvt(e1Wih, eWih1b, 768 * 256);  cvt(e1Whh, eWhh1b, 768 * 256);
  cvt(gWih, gWihb, 768 * 256);    cvt(gWhh, gWhhb, 768 * 256);
  cvt(goW, goWb, 769 * 256);
  cvt(wrW, wrWb, 257 * 256);
  cvt(crW, crWb, 256);
  k_cvt_pad<<<(256 * 288 + 255) / 256, 256, 0, stream>>>(f0W, f0Wb, 256, 257, 288);
  cvt(f1W, f1Wb, 256 * 256);
  cvt(f2W, f2Wb, 256 * 256);

  // --- conv encoder -> M0 (n,l,h) ---
  k_conv1<<<64, 256, 0, stream>>>(inp, cW1, cb1, x1);
  k_im2col<<<NL, 256, 0, stream>>>(x1, colb);
  gemmB(colb, wc2b, cb2, x2, NL, 256, 2304, 1);
  k_im2col<<<NL, 256, 0, stream>>>(x2, colb);
  gemmB(colb, wc3b, cb3, M0f, NL, 256, 2304, 1);

  // --- bi-GRU encoder: precompute gi for all positions, then 64 tiny steps ---
  gemmF(M0f, eWih0b, e0bih, giF, NL, 768, 256, 0);
  gemmF(M0f, eWih1b, e1bih, giB, NL, 768, 256, 0);
  k_zerof<<<64, 256, 0, stream>>>(henc0, 64 * 256);
  k_zerof<<<64, 256, 0, stream>>>(henc1, 64 * 256);
  for (int l = 0; l < 64; ++l) {
    gemmF(henc0, eWhh0b, e0bhh, ghE0, 64, 768, 256, 0);
    k_enc_combine<<<64, 256, 0, stream>>>(giF, ghE0, henc0, Kbuf, l, 0);
    gemmF(henc1, eWhh1b, e1bhh, ghE1, 64, 768, 256, 0);
    k_enc_combine<<<64, 256, 0, stream>>>(giB, ghE1, henc1, Kbuf, 63 - l, 1);
  }

  // --- state init: new_ep = all(rnn_hxs == 0) ---
  k_zeroint<<<1, 1, 0, stream>>>(flag);
  k_anynz<<<(NB * STATE + 255) / 256, 256, 0, stream>>>(hxs, NB * STATE, flag);
  k_selectM<<<(NB * L * H + 255) / 256, 256, 0, stream>>>(hxs, M0f, flag, Mb);
  k_copyh<<<64, 256, 0, stream>>>(hxs, h);

  // --- main scan: T=64 stream-ordered steps ---
  for (int t = 0; t < T; ++t) {
    const float* in_t = inp + (size_t)t * NB * 66;
    float* out_t = out + (size_t)t * NB * STATE;
    gemmF(h, goWb, gob, gbuf, 64, 769, 256, 0);                 // g = h@go_w.T + b
    k_attn<<<64, 256, 0, stream>>>(gbuf, Kbuf, Mb, in_t, out_t);
    k_gather<<<64, 288, 0, stream>>>(Mb, in_t, ybuf);           // y (padded to 288)
    gemmF(ybuf, f0Wb, f0b, y1, 64, 256, 288, 1);
    gemmF(y1, f1Wb, f1b, y2, 64, 256, 256, 1);
    gemmF(y2, f2Wb, f2b, y3, 64, 256, 256, 1);
    gemmF(y3, gWihb, gbih, gib, 64, 768, 256, 0);               // gi
    gemmF(h,  gWhhb, gbhh, ghb, 64, 768, 256, 0);               // gh (old h)
    k_gru_main<<<64, 256, 0, stream>>>(gib, ghb, h, out_t);     // h updated here
    gemmF(h, wrWb, wrbias, wrD, 64, 257, 256, 0);               // write head (new h)
    gemmF(h, crWb, crb, vb, 64, 1, 256, 0);                     // critic (new h)
    k_vout<<<1, 64, 0, stream>>>(vb, out_t);
    k_emitM<<<(NB * L * 64) / 256, 256, 0, stream>>>(Mb, wrD, in_t, out_t);
  }

  // hx[-1:] appended after hx
  k_lastcopy<<<(NB * STATE + 255) / 256, 256, 0, stream>>>(out);
}
